// Eq_Classifer_54022098649412
// MI455X (gfx1250) — matrix operand order repaired and checked by reading, not compiled, not run on hardware
//
#include <hip/hip_runtime.h>

#define N_NODES 131072
#define N_EDGES 2097152
#define DIM     128
#define NGRAPH  16
#define MATSZ   16384   // packed bf16 halfs per 128x128 matrix: 8 nb * 4 kb * 32 lanes * 16

typedef __attribute__((ext_vector_type(16))) __bf16 v16bf;
typedef __attribute__((ext_vector_type(8)))  float  v8f;

// ---------------- helpers ----------------

__device__ __forceinline__ unsigned short f2bf_bits(float x) {
    union { float f; unsigned u; } v; v.f = x;
    unsigned r = v.u + 0x7FFFu + ((v.u >> 16) & 1u);   // round-to-nearest-even
    return (unsigned short)(r >> 16);
}

// Load one lane's stripe of a 16x32 bf16 A-tile from a row-major f32 row.
// K in [kb*32 + kh*16, kb*32 + kh*16 + 16); caller passes k0 = kb*32 + kh*8.
__device__ __forceinline__ v16bf load_a_f32(const float* __restrict__ row, int k0) {
    v16bf a;
    const int kc = (k0 & ~31) + ((k0 >> 3) & 1) * 16;
#pragma unroll
    for (int j = 0; j < 8; ++j) a[j]     = (__bf16)row[kc + j];
#pragma unroll
    for (int j = 0; j < 8; ++j) a[8 + j] = (__bf16)row[kc + 8 + j];
    return a;
}

__device__ __forceinline__ v16bf zero_a() {
    v16bf a;
#pragma unroll
    for (int j = 0; j < 16; ++j) a[j] = (__bf16)0.0f;
    return a;
}

// ---------------- weight packing ----------------
// Packed layout per 128x128 B matrix: tile t = nb*4+kb (nb: N/16, kb: K/32),
// within tile: lane*16 + j, value = B[kb*32 + (lane>>4)*16 + j][nb*16 + (lane&15)].

__global__ void pack_conv_k(const float* __restrict__ w1, const float* __restrict__ w2,
                            const float* __restrict__ w3, unsigned short* __restrict__ out) {
    int tid = blockIdx.x * blockDim.x + threadIdx.x;      // 9 * MATSZ threads
    if (tid >= 9 * MATSZ) return;
    int mat = tid >> 14;                                   // 0..8
    int cj  = mat / 3, tap = mat % 3;
    const float* src = (cj == 0) ? w1 : (cj == 1) ? w2 : w3;
    int r    = tid & (MATSZ - 1);
    int tile = r >> 9;
    int lane = (r >> 4) & 31;
    int j    = r & 15;
    int kb = tile & 3, nb = tile >> 2;
    int k = kb * 32 + (lane >> 4) * 16 + j;                // input channel i
    int n = nb * 16 + (lane & 15);                         // output channel o
    // conv_w layout (O, I, K=3): B[k][n] = w[n][k][tap]
    out[tid] = f2bf_bits(src[(size_t)n * DIM * 3 + k * 3 + tap]);
}

__global__ void pack_g_k(const float* __restrict__ g0, const float* __restrict__ g1,
                         const float* __restrict__ g2, const float* __restrict__ g3,
                         const float* __restrict__ g4, const float* __restrict__ g5,
                         unsigned short* __restrict__ out) {
    int tid = blockIdx.x * blockDim.x + threadIdx.x;      // 6 * MATSZ threads
    if (tid >= 6 * MATSZ) return;
    int mat = tid >> 14;
    const float* src = (mat == 0) ? g0 : (mat == 1) ? g1 : (mat == 2) ? g2
                     : (mat == 3) ? g3 : (mat == 4) ? g4 : g5;
    int r    = tid & (MATSZ - 1);
    int tile = r >> 9;
    int lane = (r >> 4) & 31;
    int j    = r & 15;
    int kb = tile & 3, nb = tile >> 2;
    int k = kb * 32 + (lane >> 4) * 16 + j;
    int n = nb * 16 + (lane & 15);
    out[tid] = f2bf_bits(src[(size_t)k * DIM + n]);        // x @ W, W row-major
}

// ---------------- WMMA GEMM: out[N,128] = act(A[N,128] @ W + b) ----------------
// One wave computes a 16x128 strip. 256 threads = 8 waves = 128 rows/block.

__global__ __launch_bounds__(256) void gemm128_k(const float* __restrict__ A,
                                                 const unsigned short* __restrict__ Wp,
                                                 const float* __restrict__ bias,
                                                 float* __restrict__ out, int relu) {
    const int lane = threadIdx.x & 31;
    const int wave = threadIdx.x >> 5;
    const int row0 = (blockIdx.x * 8 + wave) * 16;
    const int m  = lane & 15;
    const int kh = lane >> 4;
    const float* arow = A + (size_t)(row0 + m) * DIM;

    v8f acc[8];
#pragma unroll
    for (int nb = 0; nb < 8; ++nb)
#pragma unroll
        for (int rr = 0; rr < 8; ++rr) acc[nb][rr] = 0.0f;

#pragma unroll
    for (int kb = 0; kb < 4; ++kb) {
        v16bf a = load_a_f32(arow, kb * 32 + kh * 8);
#pragma unroll
        for (int nb = 0; nb < 8; ++nb) {
            v16bf b = *(const v16bf*)(Wp + (((nb * 4 + kb) * 32 + lane) << 4));
            acc[nb] = __builtin_amdgcn_wmma_f32_16x16x32_bf16(
                false, a, false, b, (short)0, acc[nb], false, false);
        }
    }
    const int orow = row0 + 8 * kh;    // C layout: VGPR r -> row orow+r, col nb*16+m
#pragma unroll
    for (int nb = 0; nb < 8; ++nb) {
        float bv = bias[nb * 16 + m];
#pragma unroll
        for (int rr = 0; rr < 8; ++rr) {
            float v = acc[nb][rr] + bv;
            if (relu) v = fmaxf(v, 0.0f);
            out[(size_t)(orow + rr) * DIM + nb * 16 + m] = v;
        }
    }
}

// Dilated conv1d over the node axis as 3 row-shifted GEMMs (taps), zero padding.
__global__ __launch_bounds__(256) void conv_gemm_k(const float* __restrict__ H,
                                                   const unsigned short* __restrict__ Wp3,
                                                   const float* __restrict__ bias,
                                                   float* __restrict__ out, int dil) {
    const int lane = threadIdx.x & 31;
    const int wave = threadIdx.x >> 5;
    const int row0 = (blockIdx.x * 8 + wave) * 16;
    const int m  = lane & 15;
    const int kh = lane >> 4;

    v8f acc[8];
#pragma unroll
    for (int nb = 0; nb < 8; ++nb)
#pragma unroll
        for (int rr = 0; rr < 8; ++rr) acc[nb][rr] = 0.0f;

#pragma unroll
    for (int kb = 0; kb < 4; ++kb) {
        const int k0 = kb * 32 + kh * 8;
#pragma unroll
        for (int t = 0; t < 3; ++t) {
            int rr = row0 + m + (t - 1) * dil;
            v16bf a;
            if (rr >= 0 && rr < N_NODES) a = load_a_f32(H + (size_t)rr * DIM, k0);
            else                         a = zero_a();
            const unsigned short* W = Wp3 + t * MATSZ;
#pragma unroll
            for (int nb = 0; nb < 8; ++nb) {
                v16bf b = *(const v16bf*)(W + (((nb * 4 + kb) * 32 + lane) << 4));
                acc[nb] = __builtin_amdgcn_wmma_f32_16x16x32_bf16(
                    false, a, false, b, (short)0, acc[nb], false, false);
            }
        }
    }
    const int orow = row0 + 8 * kh;
#pragma unroll
    for (int nb = 0; nb < 8; ++nb) {
        float bv = bias[nb * 16 + m];
#pragma unroll
        for (int rr = 0; rr < 8; ++rr)
            out[(size_t)(orow + rr) * DIM + nb * 16 + m] = acc[nb][rr] + bv;
    }
}

// ---------------- elementwise / stats / edges / pooling ----------------

__global__ void zero_f32_k(float* p, long n) {
    long i = (long)blockIdx.x * blockDim.x + threadIdx.x;
    if (i < n) p[i] = 0.0f;
}

__global__ void zero4_k(float4* p, long n4) {
    long i = (long)blockIdx.x * blockDim.x + threadIdx.x;
    if (i < n4) p[i] = make_float4(0.f, 0.f, 0.f, 0.f);
}

// pre1=relu(c1)+c2, pre2=relu(c2)+c3, pre3=relu(c3)+c1 (in place) + BN sums.
__global__ __launch_bounds__(128) void fuse_stats_k(float* __restrict__ c1, float* __restrict__ c2,
                                                    float* __restrict__ c3, float* __restrict__ raw) {
    const int d = threadIdx.x;
    const long r0 = (long)blockIdx.x * 512;
    float s1 = 0, q1 = 0, s2 = 0, q2 = 0, s3 = 0, q3 = 0;
    for (int i = 0; i < 512; ++i) {
        size_t idx = (size_t)(r0 + i) * DIM + d;
        float a = c1[idx], b = c2[idx], c = c3[idx];
        float p1 = fmaxf(a, 0.f) + b;
        float p2 = fmaxf(b, 0.f) + c;
        float p3 = fmaxf(c, 0.f) + a;
        c1[idx] = p1; c2[idx] = p2; c3[idx] = p3;
        s1 += p1; q1 += p1 * p1;
        s2 += p2; q2 += p2 * p2;
        s3 += p3; q3 += p3 * p3;
    }
    atomicAdd(&raw[0 * 128 + d], s1); atomicAdd(&raw[1 * 128 + d], q1);
    atomicAdd(&raw[2 * 128 + d], s2); atomicAdd(&raw[3 * 128 + d], q2);
    atomicAdd(&raw[4 * 128 + d], s3); atomicAdd(&raw[5 * 128 + d], q3);
}

// per-channel: scale = g*rsqrt(var+eps), shift = b - mu*scale
__global__ void bn_finalize_k(const float* __restrict__ raw, float* __restrict__ ss,
                              const float* __restrict__ g, const float* __restrict__ b) {
    const int s = blockIdx.x, d = threadIdx.x;
    const float invN = 1.0f / (float)N_NODES;
    float mu  = raw[(2 * s) * 128 + d] * invN;
    float var = raw[(2 * s + 1) * 128 + d] * invN - mu * mu;
    float sc  = g[d] * rsqrtf(var + 1e-5f);
    ss[(2 * s) * 128 + d]     = sc;
    ss[(2 * s + 1) * 128 + d] = b[d] - mu * sc;
}

__global__ void normalize3_k(float* __restrict__ b0, float* __restrict__ b1,
                             float* __restrict__ b2, const float* __restrict__ ss) {
    const long n = (long)N_NODES * DIM;
    for (long i = (long)blockIdx.x * blockDim.x + threadIdx.x; i < n;
         i += (long)gridDim.x * blockDim.x) {
        int d = (int)(i & 127);
        b0[i] = b0[i] * ss[d]       + ss[128 + d];
        b1[i] = b1[i] * ss[256 + d] + ss[384 + d];
        b2[i] = b2[i] * ss[512 + d] + ss[640 + d];
    }
}

__global__ void normalize1_k(float* __restrict__ b0, const float* __restrict__ ss) {
    const long n = (long)N_NODES * DIM;
    for (long i = (long)blockIdx.x * blockDim.x + threadIdx.x; i < n;
         i += (long)gridDim.x * blockDim.x) {
        int d = (int)(i & 127);
        b0[i] = b0[i] * ss[d] + ss[128 + d];
    }
}

// agg[dst] += X[src] * ew  (one wave per edge; lane c handles 4 floats)
__global__ __launch_bounds__(256) void aggregate_k(const float* __restrict__ X,
                                                   const int* __restrict__ src,
                                                   const int* __restrict__ dst,
                                                   const float* __restrict__ ew,
                                                   float* __restrict__ agg) {
    long idx = (long)blockIdx.x * blockDim.x + threadIdx.x;
    long e = idx >> 5;
    int  c = (int)(idx & 31);
    if (e >= N_EDGES) return;
    int s = src[e], d = dst[e];
    float w = ew[e];
    const float4 v = *(const float4*)(X + (size_t)s * DIM + c * 4);
    float* p = agg + (size_t)d * DIM + c * 4;
    atomicAdd(p + 0, v.x * w);
    atomicAdd(p + 1, v.y * w);
    atomicAdd(p + 2, v.z * w);
    atomicAdd(p + 3, v.w * w);
}

// press: out[n,d] = sum_c sum_k a_c[n, d+k-1]*pw[c,k] + pb  + BN sums
__global__ __launch_bounds__(128) void press_stats_k(const float* __restrict__ a1,
                                                     const float* __restrict__ a2,
                                                     const float* __restrict__ a3,
                                                     const float* __restrict__ pw,
                                                     const float* __restrict__ pb,
                                                     float* __restrict__ out,
                                                     float* __restrict__ raw) {
    const int d = threadIdx.x;
    const long r0 = (long)blockIdx.x * 512;
    float w[9];
#pragma unroll
    for (int i = 0; i < 9; ++i) w[i] = pw[i];
    const float bias = pb[0];
    float s = 0, q = 0;
    for (int i = 0; i < 512; ++i) {
        size_t base = (size_t)(r0 + i) * DIM;
        float acc = bias;
        const float* chan[3] = { a1 + base, a2 + base, a3 + base };
#pragma unroll
        for (int c = 0; c < 3; ++c) {
            float xm = (d > 0)       ? chan[c][d - 1] : 0.0f;
            float x0 = chan[c][d];
            float xp = (d < DIM - 1) ? chan[c][d + 1] : 0.0f;
            acc += xm * w[c * 3 + 0] + x0 * w[c * 3 + 1] + xp * w[c * 3 + 2];
        }
        out[base + d] = acc;
        s += acc; q += acc * acc;
    }
    atomicAdd(&raw[d], s);
    atomicAdd(&raw[128 + d], q);
}

// graph pooling: hg[gid[n]][col_off + d] += X[n][d], LDS-staged
__global__ __launch_bounds__(256) void segsum_k(const float* __restrict__ X,
                                                const int* __restrict__ gid,
                                                float* __restrict__ hg, int col_off) {
    __shared__ float acc[NGRAPH * DIM];
    for (int i = threadIdx.x; i < NGRAPH * DIM; i += 256) acc[i] = 0.0f;
    __syncthreads();
    const int c = threadIdx.x & 127;
    const int rh = threadIdx.x >> 7;                 // 0 or 1
    const long r0 = (long)blockIdx.x * 512;
    for (int i = rh; i < 512; i += 2) {
        long r = r0 + i;
        int g = gid[r];
        atomicAdd(&acc[g * DIM + c], X[(size_t)r * DIM + c]);
    }
    __syncthreads();
    for (int i = threadIdx.x; i < NGRAPH * DIM; i += 256)
        atomicAdd(&hg[(i >> 7) * (3 * DIM) + col_off + (i & 127)], acc[i]);
}

// out[16,5] = (hg[16,384] @ cls1_w.T + b1) @ cls2_w.T + b2
__global__ __launch_bounds__(128) void classifier_k(const float* __restrict__ hg,
                                                    const float* __restrict__ w1,
                                                    const float* __restrict__ b1,
                                                    const float* __restrict__ w2,
                                                    const float* __restrict__ b2,
                                                    float* __restrict__ out) {
    __shared__ float t[NGRAPH * DIM];
    const int o = threadIdx.x;
    for (int g = 0; g < NGRAPH; ++g) {
        float acc = b1[o];
        const float* hr = hg + g * 3 * DIM;
        const float* wr = w1 + (size_t)o * 3 * DIM;
        for (int k = 0; k < 3 * DIM; ++k) acc += hr[k] * wr[k];
        t[g * DIM + o] = acc;
    }
    __syncthreads();
    if (threadIdx.x < NGRAPH * 5) {
        int g = threadIdx.x / 5, c = threadIdx.x % 5;
        float acc = b2[c];
        const float* tr = t + g * DIM;
        const float* wr = w2 + c * DIM;
        for (int k = 0; k < DIM; ++k) acc += tr[k] * wr[k];
        out[g * 5 + c] = acc;
    }
}

// ---------------- host orchestration ----------------

extern "C" void kernel_launch(void* const* d_in, const int* in_sizes, int n_in,
                              void* d_out, int out_size, void* d_ws, size_t ws_size,
                              hipStream_t stream) {
    (void)in_sizes; (void)n_in; (void)out_size; (void)ws_size;

    const float* h      = (const float*)d_in[0];
    const float* edge_w = (const float*)d_in[1];
    const int*   src    = (const int*)d_in[2];
    const int*   dst    = (const int*)d_in[3];
    const int*   gid    = (const int*)d_in[4];
    const float* cw[3]  = { (const float*)d_in[6], (const float*)d_in[8], (const float*)d_in[10] };
    const float* cb[3]  = { (const float*)d_in[7], (const float*)d_in[9], (const float*)d_in[11] };
    const float* press_w = (const float*)d_in[12];
    const float* press_b = (const float*)d_in[13];
    const float* gw[6] = { (const float*)d_in[14], (const float*)d_in[16], (const float*)d_in[18],
                           (const float*)d_in[20], (const float*)d_in[22], (const float*)d_in[24] };
    const float* gb[6] = { (const float*)d_in[15], (const float*)d_in[17], (const float*)d_in[19],
                           (const float*)d_in[21], (const float*)d_in[23], (const float*)d_in[25] };
    const float* bn_g   = (const float*)d_in[26];
    const float* bn_b   = (const float*)d_in[27];
    const float* cls1_w = (const float*)d_in[28];
    const float* cls1_b = (const float*)d_in[29];
    const float* cls2_w = (const float*)d_in[30];
    const float* cls2_b = (const float*)d_in[31];
    float* out = (float*)d_out;

    char* ws = (char*)d_ws;
    const size_t NB = (size_t)N_NODES * DIM * sizeof(float);   // 64 MiB
    float* B[7];
    for (int i = 0; i < 7; ++i) B[i] = (float*)(ws + (size_t)i * NB);
    unsigned short* PW = (unsigned short*)(ws + 7 * NB);       // 15 packed matrices
    float* SB = (float*)(ws + 7 * NB + (size_t)15 * MATSZ * sizeof(unsigned short));
    float* RAW1 = SB;            // 6*128: conv-stage BN sums
    float* SS1  = SB + 768;      // 6*128: conv-stage scale/shift
    float* RAWP = SB + 1536;     // 2*128: press-stage sums
    float* SSP  = SB + 1792;     // 2*128: press-stage scale/shift
    float* HG   = SB + 4096;     // 16 x 384 pooled features

    const long ND  = (long)N_NODES * DIM;
    const int GEMM_GRID = N_NODES / 128;                 // 1024 blocks, 256 thr
    const int AGG_GRID  = (int)(((long)N_EDGES * 32) / 256);
    const int Z4_GRID   = (int)((ND / 4 + 255) / 256);

    // 0. zero stats + pooled features
    zero_f32_k<<<(4096 + NGRAPH * 3 * DIM + 255) / 256, 256, 0, stream>>>(SB, 4096 + NGRAPH * 3 * DIM);

    // 1. pack weights to WMMA B-layout (bf16)
    pack_conv_k<<<(9 * MATSZ + 255) / 256, 256, 0, stream>>>(cw[0], cw[1], cw[2], PW);
    pack_g_k<<<(6 * MATSZ + 255) / 256, 256, 0, stream>>>(gw[0], gw[1], gw[2], gw[3], gw[4], gw[5],
                                                          PW + 9 * MATSZ);

    // 2. dilated convs c1,c2,c3 -> B0,B1,B2
    for (int j = 0; j < 3; ++j)
        conv_gemm_k<<<GEMM_GRID, 256, 0, stream>>>(h, PW + j * 3 * MATSZ, cb[j], B[j], j + 1);

    // 3. fused pre-activations (in place) + BN stats; finalize; normalize -> h01..h03
    fuse_stats_k<<<N_NODES / 512, 128, 0, stream>>>(B[0], B[1], B[2], RAW1);
    bn_finalize_k<<<3, 128, 0, stream>>>(RAW1, SS1, bn_g, bn_b);
    normalize3_k<<<4096, 256, 0, stream>>>(B[0], B[1], B[2], SS1);

    // 4. first three graph convs: a_j = relu(agg(h0j) @ g1j + b) -> B3,B4,B5
    for (int j = 0; j < 3; ++j) {
        zero4_k<<<Z4_GRID, 256, 0, stream>>>((float4*)B[6], ND / 4);
        aggregate_k<<<AGG_GRID, 256, 0, stream>>>(B[j], src, dst, edge_w, B[6]);
        gemm128_k<<<GEMM_GRID, 256, 0, stream>>>(B[6], PW + (9 + j) * MATSZ, gb[j], B[3 + j], 1);
    }

    // 5. press conv over (a1,a2,a3) -> B0, + BN -> ac_h1
    press_stats_k<<<N_NODES / 512, 128, 0, stream>>>(B[3], B[4], B[5], press_w, press_b, B[0], RAWP);
    bn_finalize_k<<<1, 128, 0, stream>>>(RAWP, SSP, bn_g, bn_b);
    normalize1_k<<<4096, 256, 0, stream>>>(B[0], SSP);

    // 6. gconv chain + graph pooling
    // ac_h2 = relu(agg(ac_h1) @ g2) -> B1 ; hg[:,0:128]
    zero4_k<<<Z4_GRID, 256, 0, stream>>>((float4*)B[6], ND / 4);
    aggregate_k<<<AGG_GRID, 256, 0, stream>>>(B[0], src, dst, edge_w, B[6]);
    gemm128_k<<<GEMM_GRID, 256, 0, stream>>>(B[6], PW + 12 * MATSZ, gb[3], B[1], 1);
    segsum_k<<<N_NODES / 512, 256, 0, stream>>>(B[1], gid, HG, 0);
    // ac_h3 -> B0 ; hg[:,128:256]
    zero4_k<<<Z4_GRID, 256, 0, stream>>>((float4*)B[6], ND / 4);
    aggregate_k<<<AGG_GRID, 256, 0, stream>>>(B[1], src, dst, edge_w, B[6]);
    gemm128_k<<<GEMM_GRID, 256, 0, stream>>>(B[6], PW + 13 * MATSZ, gb[4], B[0], 1);
    segsum_k<<<N_NODES / 512, 256, 0, stream>>>(B[0], gid, HG, 128);
    // ac_h4 -> B1 ; hg[:,256:384]
    zero4_k<<<Z4_GRID, 256, 0, stream>>>((float4*)B[6], ND / 4);
    aggregate_k<<<AGG_GRID, 256, 0, stream>>>(B[0], src, dst, edge_w, B[6]);
    gemm128_k<<<GEMM_GRID, 256, 0, stream>>>(B[6], PW + 14 * MATSZ, gb[5], B[1], 1);
    segsum_k<<<N_NODES / 512, 256, 0, stream>>>(B[1], gid, HG, 256);

    // 7. classifier head
    classifier_k<<<1, 128, 0, stream>>>(HG, cls1_w, cls1_b, cls2_w, cls2_b, out);
}
